// SpectralEncoder_6545530159339
// MI455X (gfx1250) — compile-verified, matrix operand order checked by reading, not run on hardware
//
#include <hip/hip_runtime.h>

typedef __attribute__((ext_vector_type(16))) _Float16 v16h;
typedef __attribute__((ext_vector_type(8)))  _Float16 v8h;
typedef __attribute__((ext_vector_type(8)))  float    v8f;
typedef __attribute__((ext_vector_type(4)))  float    v4f;
typedef __attribute__((ext_vector_type(4)))  unsigned int u32x4;
typedef __attribute__((ext_vector_type(8)))  int      i32x8;
typedef __attribute__((ext_vector_type(4)))  int      i32x4;

#define IN_DIM 128
#define PE_DIM 16
#define F_REAL 144      // IN + PE
#define F1P    160      // padded to 5*32 for WMMA K-loop
#define HID    128
#define LAT    64
#define KCH    4

#if defined(__HIP_DEVICE_COMPILE__) && __has_builtin(__builtin_amdgcn_tensor_load_to_lds)
#define USE_TDM 1
#endif

// ---------------- utility kernels ----------------

__global__ void fill_zero_f32(float* __restrict__ p, long long n) {
  long long i = (long long)blockIdx.x * blockDim.x + threadIdx.x;
  long long stride = (long long)gridDim.x * blockDim.x;
  for (; i < n; i += stride) p[i] = 0.f;
}

__global__ void deg_scatter_k(const int* __restrict__ src, const int* __restrict__ dst,
                              const float* __restrict__ ew, float* __restrict__ deg, int E) {
  int e = blockIdx.x * blockDim.x + threadIdx.x;
  if (e >= E) return;
  int s = src[e];
  float w = (s == dst[e]) ? 0.f : ew[e];
  if (w != 0.f) atomicAdd(&deg[s], w);
}

__global__ void dinv_k(const float* __restrict__ deg, float* __restrict__ dinv, int n) {
  int i = blockIdx.x * blockDim.x + threadIdx.x;
  if (i >= n) return;
  float d = deg[i];
  dinv[i] = d > 0.f ? rsqrtf(d) : 0.f;
}

__global__ void what_k(const int* __restrict__ src, const int* __restrict__ dst,
                       const float* __restrict__ ew, const float* __restrict__ dinv,
                       float* __restrict__ what, int E) {
  int e = blockIdx.x * blockDim.x + threadIdx.x;
  if (e >= E) return;
  int s = src[e], d = dst[e];
  float w = (s == d) ? 0.f : ew[e];
  what[e] = -dinv[s] * w * dinv[d];
}

// h0 = concat(x, lap_pe) padded to F1P, as f16. blockDim = F1P, grid = N
__global__ void build_h0_k(const float* __restrict__ x, const float* __restrict__ pe,
                           _Float16* __restrict__ h0) {
  int i = blockIdx.x;
  int f = threadIdx.x;
  float v;
  if (f < IN_DIM)       v = x[(size_t)i * IN_DIM + f];
  else if (f < F_REAL)  v = pe[(size_t)i * PE_DIM + (f - IN_DIM)];
  else                  v = 0.f;
  h0[(size_t)i * F1P + f] = (_Float16)v;
}

// W1 [K,144,128] f32 -> W1T [K,128,160] f16 (transposed, K-padded with zeros)
__global__ void conv_w1_k(const float* __restrict__ W, _Float16* __restrict__ WT) {
  int t = blockIdx.x * blockDim.x + threadIdx.x;
  if (t >= KCH * HID * F1P) return;
  int k  = t / (HID * F1P);
  int r  = t - k * (HID * F1P);
  int n  = r / F1P;
  int kk = r - n * F1P;
  float v = (kk < F_REAL) ? W[((size_t)k * F_REAL + kk) * HID + n] : 0.f;
  WT[t] = (_Float16)v;
}

// W2 [K,128,128] f32 -> W2T [K,128,128] f16 (transposed)
__global__ void conv_w2_k(const float* __restrict__ W, _Float16* __restrict__ WT) {
  int t = blockIdx.x * blockDim.x + threadIdx.x;
  if (t >= KCH * HID * HID) return;
  int k  = t / (HID * HID);
  int r  = t - k * (HID * HID);
  int n  = r / HID;
  int kk = r - n * HID;
  WT[t] = (_Float16)W[((size_t)k * HID + kk) * HID + n];
}

__global__ void init_out_k(float* __restrict__ out, const float* __restrict__ b, long long n) {
  long long i = (long long)blockIdx.x * blockDim.x + threadIdx.x;
  if (i >= n) return;
  out[i] = b[(int)(i & (HID - 1))];
}

// ---------------- WMMA GEMM with TDM-staged B tile ----------------
// C[N,128] += A[N,K](f16) * BT[128,K](f16)^T.
// 8 waves/block; wave w -> 16x16 tile at cols [16w,16w+16); block -> 16 rows.
// Whole BT (128*K f16) is DMA'd into LDS once per block by the Tensor Data Mover
// with padding: 4 dwords inserted every 16 dwords -> LDS row stride = K + K/4 halfs.
// Padded in-row half offset for half index h:  h + 8*(h>>5)  (keeps 16B alignment).
__global__ __launch_bounds__(256)
void gemm_wmma_k(const _Float16* __restrict__ A, const _Float16* __restrict__ BT,
                 float* __restrict__ C, int K, int Nrows) {
  extern __shared__ _Float16 Bs[];
  const int strideH = K + (K >> 2);              // padded LDS row stride in halfs

#if defined(USE_TDM)
  if (threadIdx.x < 32) {                        // wave 0 issues the TDM descriptor
    unsigned lds_base = (unsigned)(size_t)(&Bs[0]);   // low 32 bits of flat = LDS offset
    unsigned long long ga = (unsigned long long)(size_t)BT;
    u32x4 g0;
    g0[0] = 1u;                                   // count=1, user mode
    g0[1] = lds_base;                             // lds_addr
    g0[2] = (unsigned)(ga & 0xffffffffu);         // global_addr[31:0]
    g0[3] = (unsigned)((ga >> 32) & 0x01ffffffu)  // global_addr[56:32]
          | 0x80000000u;                          // type=2 (image)
    i32x8 g1;
    g1[0] = (int)0x06D10000;  // data_size=2B, pad_enable, pad_interval=16dw, pad_amount=4dw
    g1[1] = (int)((K & 0xffff) << 16);            // tensor_dim0[15:0]
    g1[2] = (int)(HID << 16);                     // tensor_dim0[31:16]=0 | tensor_dim1[15:0]=128
    g1[3] = (int)(K << 16);                       // tensor_dim1[31:16]=0 | tile_dim0=K
    g1[4] = HID;                                  // tile_dim1=128, tile_dim2=0
    g1[5] = K;                                    // tensor_dim0_stride[31:0]
    g1[6] = 0;                                    // stride hi, tensor_dim1_stride lo
    g1[7] = 0;
    i32x4 z4 = {0, 0, 0, 0};
#if __clang_major__ >= 23
    i32x8 z8 = {0, 0, 0, 0, 0, 0, 0, 0};
    __builtin_amdgcn_tensor_load_to_lds(g0, g1, z4, z4, z8, 0);
#else
    __builtin_amdgcn_tensor_load_to_lds(g0, g1, z4, z4, 0);
#endif
    __builtin_amdgcn_s_wait_tensorcnt((short)0);
  }
#elif defined(__HIP_DEVICE_COMPILE__)
#error "CDNA5 probe: tensor_load_to_lds builtin NOT available - TDM path was never compiled"
#else
  // host compilation pass only (never codegen'd): cooperative fill, same padded layout
  for (int t = threadIdx.x; t < HID * K; t += 256) {
    int r = t / K;
    int h = t - r * K;
    Bs[r * strideH + h + ((h >> 5) << 3)] = BT[(size_t)r * K + h];
  }
#endif
  __syncthreads();

  const int wave  = threadIdx.x >> 5;
  const int lane  = threadIdx.x & 31;
  const int hsel  = lane >> 4;                   // which 16-wide K-chunk this lane carries
  const int l     = lane & 15;
  const int mbase = blockIdx.x * 16;
  const int cbase = wave * 16;

  int arow = mbase + l;
  if (arow >= Nrows) arow = Nrows - 1;           // tail clamp (writes guarded below)
  const _Float16* ap = A + (size_t)arow * K + hsel * 16;
  const _Float16* bp = Bs + (size_t)(cbase + l) * strideH;

  v8f acc = {};
  for (int kb = 0; kb < K; kb += 32) {
    int h = kb + hsel * 16;
    v16h a = *(const v16h*)(ap + kb);
    v16h b = *(const v16h*)(bp + h + ((h >> 5) << 3));
    acc = __builtin_amdgcn_wmma_f32_16x16x32_f16(
        /*neg_a=*/false, a, /*neg_b=*/false, b,
        /*c_mod=*/(short)0, acc, /*reuse_a=*/false, /*reuse_b=*/false);
  }

  // C/D layout: VGPR r, lanes 0-15 -> M = r, lanes 16-31 -> M = 8 + r ; N = cbase + l
  const int col = cbase + l;
#pragma unroll
  for (int r = 0; r < 8; ++r) {
    int row = mbase + r + 8 * hsel;
    if (row < Nrows) {
      float* cp = C + (size_t)row * HID + col;
      *cp += acc[r];
    }
  }
}

// ---------------- SpMM hop: P[dst] += w_hat * T[src] ----------------
// block = (F/8, 8): threadIdx.y = edge-in-block, threadIdx.x = 8-feature chunk.
// P must be pre-zeroed; the combine kernels re-zero it for the next hop.
__global__ void spmm_scatter_k(const int* __restrict__ src, const int* __restrict__ dst,
                               const float* __restrict__ what, const _Float16* __restrict__ T,
                               float* __restrict__ P, int F, int E) {
  int e = blockIdx.x * blockDim.y + threadIdx.y;
  if (e >= E) return;
  float w = what[e];
  if (w == 0.f) return;
  int s = src[e];
  int d = dst[e];
  int c = threadIdx.x << 3;
  v8h hv = *(const v8h*)(T + (size_t)s * F + c);
  float* pp = P + (size_t)d * F + c;
#pragma unroll
  for (int j = 0; j < 8; ++j) atomicAdd(pp + j, w * (float)hv[j]);
}

// T <- (f16)P ; P <- 0   (first hop), 8 elements per thread
__global__ void combine_copy_k(float* __restrict__ P, _Float16* __restrict__ T, long long n8) {
  long long i = (long long)blockIdx.x * blockDim.x + threadIdx.x;
  if (i >= n8) return;
  v4f* p = (v4f*)P + 2 * i;
  v4f a = p[0], b = p[1];
  v8h o;
#pragma unroll
  for (int j = 0; j < 4; ++j) { o[j] = (_Float16)a[j]; o[4 + j] = (_Float16)b[j]; }
  ((v8h*)T)[i] = o;
  v4f z = {0.f, 0.f, 0.f, 0.f};
  p[0] = z;
  p[1] = z;
}

// T <- (f16)(2*P - T) ; P <- 0   (Chebyshev recurrence, overwrite T_{k-2} slot)
__global__ void combine_rec_k(float* __restrict__ P, _Float16* __restrict__ T, long long n8) {
  long long i = (long long)blockIdx.x * blockDim.x + threadIdx.x;
  if (i >= n8) return;
  v4f* p = (v4f*)P + 2 * i;
  v4f a = p[0], b = p[1];
  v8h t = ((const v8h*)T)[i];
  v8h o;
#pragma unroll
  for (int j = 0; j < 4; ++j) {
    o[j]     = (_Float16)(2.f * a[j] - (float)t[j]);
    o[4 + j] = (_Float16)(2.f * b[j] - (float)t[4 + j]);
  }
  ((v8h*)T)[i] = o;
  v4f z = {0.f, 0.f, 0.f, 0.f};
  p[0] = z;
  p[1] = z;
}

__global__ void relu_f16_k(const float* __restrict__ out, _Float16* __restrict__ T, long long n8) {
  long long i = (long long)blockIdx.x * blockDim.x + threadIdx.x;
  if (i >= n8) return;
  const v4f* p = (const v4f*)out + 2 * i;
  v4f a = p[0], b = p[1];
  v8h o;
#pragma unroll
  for (int j = 0; j < 4; ++j) {
    o[j]     = (_Float16)fmaxf(a[j], 0.f);
    o[4 + j] = (_Float16)fmaxf(b[j], 0.f);
  }
  ((v8h*)T)[i] = o;
}

// g[c] = mean_i relu(out[i,c]) ; one block per column
__global__ void colmean_relu_k(const float* __restrict__ out, float* __restrict__ g, int n) {
  __shared__ float s[256];
  int c = blockIdx.x;
  float acc = 0.f;
  for (int i = threadIdx.x; i < n; i += blockDim.x)
    acc += fmaxf(out[(size_t)i * HID + c], 0.f);
  s[threadIdx.x] = acc;
  __syncthreads();
  for (int o = 128; o > 0; o >>= 1) {
    if (threadIdx.x < o) s[threadIdx.x] += s[threadIdx.x + o];
    __syncthreads();
  }
  if (threadIdx.x == 0) g[c] = s[0] / (float)n;
}

// out[0:64]=g@W_mu+b_mu ; out[64:128]=g@W_lv+b_lv ; one block of 128 threads
__global__ void head_k(const float* __restrict__ g,
                       const float* __restrict__ Wmu, const float* __restrict__ bmu,
                       const float* __restrict__ Wlv, const float* __restrict__ blv,
                       float* __restrict__ o) {
  __shared__ float gs[HID];
  int t = threadIdx.x;
  gs[t] = g[t];
  __syncthreads();
  if (t < LAT) {
    float a = bmu[t];
    for (int c = 0; c < HID; ++c) a += gs[c] * Wmu[c * LAT + t];
    o[t] = a;
  } else {
    int j = t - LAT;
    float a = blv[j];
    for (int c = 0; c < HID; ++c) a += gs[c] * Wlv[c * LAT + j];
    o[LAT + j] = a;
  }
}

// ---------------- host launch ----------------

extern "C" void kernel_launch(void* const* d_in, const int* in_sizes, int n_in,
                              void* d_out, int out_size, void* d_ws, size_t ws_size,
                              hipStream_t stream) {
  const float* x    = (const float*)d_in[0];
  const int*   ei   = (const int*)  d_in[1];
  const float* pe   = (const float*)d_in[2];
  const float* ew   = (const float*)d_in[3];
  const float* W1   = (const float*)d_in[4];
  const float* b1   = (const float*)d_in[5];
  const float* W2   = (const float*)d_in[6];
  const float* b2   = (const float*)d_in[7];
  const float* Wmu  = (const float*)d_in[8];
  const float* bmu  = (const float*)d_in[9];
  const float* Wlv  = (const float*)d_in[10];
  const float* blv  = (const float*)d_in[11];
  float* out_final  = (float*)d_out;

  const int N = in_sizes[0] / IN_DIM;
  const int E = in_sizes[3];
  const int* src = ei;
  const int* dst = ei + E;

  // workspace carve-out (256B aligned partitions)
  char* base = (char*)d_ws;
  size_t off = 0;
  auto carve = [&](size_t bytes) -> void* {
    void* p = base + off;
    off += (bytes + 255) & ~(size_t)255;
    return p;
  };
  float*    deg  = (float*)   carve((size_t)N * 4);
  float*    dinv = (float*)   carve((size_t)N * 4);
  float*    what = (float*)   carve((size_t)E * 4);
  _Float16* W1T  = (_Float16*)carve((size_t)KCH * HID * F1P * 2);
  _Float16* W2T  = (_Float16*)carve((size_t)KCH * HID * HID * 2);
  _Float16* TX0  = (_Float16*)carve((size_t)N * F1P * 2);
  _Float16* TX1  = (_Float16*)carve((size_t)N * F1P * 2);
  float*    P    = (float*)   carve((size_t)N * F1P * 4);
  float*    OUT  = (float*)   carve((size_t)N * HID * 4);
  float*    G    = (float*)   carve((size_t)HID * 4);
  (void)ws_size; (void)n_in; (void)out_size;

  const int TB = 256;
  const long long nNH = (long long)N * HID;
  const int gemmGrid  = (N + 15) / 16;
  const size_t shB1   = (size_t)(F1P + F1P / 4) * HID * 2;  // padded LDS B tile, K=160
  const size_t shB2   = (size_t)(HID + HID / 4) * HID * 2;  // padded LDS B tile, K=128

  auto cdiv = [](long long a, int b) { return (int)((a + b - 1) / b); };

  // ---- preprocess ----
  fill_zero_f32<<<cdiv(N, TB), TB, 0, stream>>>(deg, N);
  fill_zero_f32<<<cdiv((long long)N * F1P, TB), TB, 0, stream>>>(P, (long long)N * F1P);
  deg_scatter_k<<<cdiv(E, TB), TB, 0, stream>>>(src, dst, ew, deg, E);
  dinv_k<<<cdiv(N, TB), TB, 0, stream>>>(deg, dinv, N);
  what_k<<<cdiv(E, TB), TB, 0, stream>>>(src, dst, ew, dinv, what, E);
  conv_w1_k<<<cdiv((long long)KCH * HID * F1P, TB), TB, 0, stream>>>(W1, W1T);
  conv_w2_k<<<cdiv((long long)KCH * HID * HID, TB), TB, 0, stream>>>(W2, W2T);
  build_h0_k<<<N, F1P, 0, stream>>>(x, pe, TX0);

  // one SpMM hop: P = L_hat * Tin (features F), combine into Tdst, P re-zeroed by combine
  auto hop = [&](const _Float16* Tin, _Float16* Tdst, int F, bool firstHop) {
    long long nNF = (long long)N * F;
    dim3 sb(F / 8, 8);
    spmm_scatter_k<<<cdiv(E, 8), sb, 0, stream>>>(src, dst, what, Tin, P, F, E);
    if (firstHop)
      combine_copy_k<<<cdiv(nNF / 8, TB), TB, 0, stream>>>(P, Tdst, nNF / 8);
    else
      combine_rec_k<<<cdiv(nNF / 8, TB), TB, 0, stream>>>(P, Tdst, nNF / 8);
  };

  // ---- layer 1 (K = F1P) ----
  init_out_k<<<cdiv(nNH, TB), TB, 0, stream>>>(OUT, b1, nNH);
  gemm_wmma_k<<<gemmGrid, 256, shB1, stream>>>(TX0, W1T + 0 * HID * F1P, OUT, F1P, N);
  hop(TX0, TX1, F1P, true);                                   // T1 = L_hat T0
  gemm_wmma_k<<<gemmGrid, 256, shB1, stream>>>(TX1, W1T + 1 * HID * F1P, OUT, F1P, N);
  hop(TX1, TX0, F1P, false);                                  // T2 = 2 L_hat T1 - T0
  gemm_wmma_k<<<gemmGrid, 256, shB1, stream>>>(TX0, W1T + 2 * HID * F1P, OUT, F1P, N);
  hop(TX0, TX1, F1P, false);                                  // T3 = 2 L_hat T2 - T1
  gemm_wmma_k<<<gemmGrid, 256, shB1, stream>>>(TX1, W1T + 3 * HID * F1P, OUT, F1P, N);
  relu_f16_k<<<cdiv(nNH / 8, TB), TB, 0, stream>>>(OUT, TX0, nNH / 8); // h1 (f16, stride HID)

  // ---- layer 2 (K = HID) ----
  init_out_k<<<cdiv(nNH, TB), TB, 0, stream>>>(OUT, b2, nNH);
  gemm_wmma_k<<<gemmGrid, 256, shB2, stream>>>(TX0, W2T + 0 * HID * HID, OUT, HID, N);
  hop(TX0, TX1, HID, true);
  gemm_wmma_k<<<gemmGrid, 256, shB2, stream>>>(TX1, W2T + 1 * HID * HID, OUT, HID, N);
  hop(TX1, TX0, HID, false);
  gemm_wmma_k<<<gemmGrid, 256, shB2, stream>>>(TX0, W2T + 2 * HID * HID, OUT, HID, N);
  hop(TX0, TX1, HID, false);
  gemm_wmma_k<<<gemmGrid, 256, shB2, stream>>>(TX1, W2T + 3 * HID * HID, OUT, HID, N);

  // ---- pool + heads ----
  colmean_relu_k<<<HID, 256, 0, stream>>>(OUT, G, N);
  head_k<<<1, HID, 0, stream>>>(G, Wmu, bmu, Wlv, blv, out_final);
}
